// TransformerBlock_82300163326106
// MI455X (gfx1250) — matrix operand order repaired and checked by reading, not compiled, not run on hardware
//
#include <hip/hip_runtime.h>

// ---------------------------------------------------------------------------
// MI455X transformer block: bf16 WMMA (v_wmma_f32_16x16x32_bf16, f32 accum)
// for all 7 GEMMs + both attention matmuls; async global->LDS DMA
// (global_load_async_to_lds_b128 / s_wait_asynccnt) for tile staging with
// double-buffered LDS so DMA overlaps matrix work. Compute-bound
// (~2000 FLOP/B), so everything funnels into the matrix pipe.
// ---------------------------------------------------------------------------

typedef __attribute__((ext_vector_type(16))) __bf16 v16bf;
typedef __attribute__((ext_vector_type(8)))  float  v8f;

#define D_MODEL 1024
#define N_HEADS 16
#define DKH     64
#define D_FF    4096
#define BSZ     2
#define SEQ     2048
#define NTOK    (BSZ * SEQ)   // 4096

typedef unsigned short ushort_t;

static __device__ __forceinline__ ushort_t f2bf(float f) {
  union { float f; unsigned u; } v; v.f = f;
  unsigned r = v.u + 0x7FFFu + ((v.u >> 16) & 1u);   // round-to-nearest-even
  return (ushort_t)(r >> 16);
}
static __device__ __forceinline__ float bf2f(ushort_t h) {
  union { float f; unsigned u; } v; v.u = ((unsigned)h) << 16; return v.f;
}

union Frag16 { v16bf v; unsigned u[8]; };

// A-fragment k-offset pattern for 16x32 bf16 A: contiguous, k = 16*half + 2j (matches B fills).
static __device__ __forceinline__ int kmapA(int j, int half) {
  return 2 * j + 16 * half;
}

// ---------------------------------------------------------------------------
// CDNA5 async DMA: global -> LDS, 16B per lane, tracked by ASYNCcnt.
// LDS byte address = low 32 bits of the flat pointer (ISA 10.2 LDS aperture).
// ---------------------------------------------------------------------------
static __device__ __forceinline__ void async_load_b128(const void* gsrc, void* lds_dst) {
  unsigned lds = (unsigned)(unsigned long long)(uintptr_t)lds_dst;
  asm volatile("global_load_async_to_lds_b128 %0, %1, off"
               :: "v"(lds), "v"(gsrc)
               : "memory");
}
static __device__ __forceinline__ void wait_async0() {
  asm volatile("s_wait_asynccnt 0x0" ::: "memory");
}

// ---------------------------------------------------------------------------
// f32 -> bf16 conversion (weights)
// ---------------------------------------------------------------------------
__global__ void cvt_bf16_kernel(const float* __restrict__ src,
                                ushort_t* __restrict__ dst, int n) {
  int i = blockIdx.x * blockDim.x + threadIdx.x;
  if (i < n) dst[i] = f2bf(src[i]);
}

// ---------------------------------------------------------------------------
// RMSNorm: one 256-thread block per token row of 1024
// ---------------------------------------------------------------------------
__global__ __launch_bounds__(256)
void rmsnorm_kernel(const float* __restrict__ x, const float* __restrict__ g,
                    ushort_t* __restrict__ out) {
  __shared__ float red[8];
  __shared__ float sscale;
  const int row = blockIdx.x;
  const int tid = threadIdx.x;
  const float4 v = *(const float4*)(x + (size_t)row * D_MODEL + tid * 4);
  float s = v.x * v.x + v.y * v.y + v.z * v.z + v.w * v.w;
#pragma unroll
  for (int off = 16; off; off >>= 1) s += __shfl_xor(s, off, 32);
  if ((tid & 31) == 0) red[tid >> 5] = s;
  __syncthreads();
  if (tid == 0) {
    float t = 0.f;
#pragma unroll
    for (int i = 0; i < 8; ++i) t += red[i];
    sscale = rsqrtf(t * (1.0f / D_MODEL) + 1e-5f);
  }
  __syncthreads();
  const float sc = sscale;
  const float4 gg = *(const float4*)(g + tid * 4);
  size_t o = (size_t)row * D_MODEL + tid * 4;
  out[o + 0] = f2bf(v.x * sc * gg.x);
  out[o + 1] = f2bf(v.y * sc * gg.y);
  out[o + 2] = f2bf(v.z * sc * gg.z);
  out[o + 3] = f2bf(v.w * sc * gg.w);
}

// ---------------------------------------------------------------------------
// Tiled bf16 WMMA GEMM:  C[M,N] = A[M,K] * W[N,K]^T (+resid), out f32 or bf16
// Block tile 128x128, k-step 32; 8 waves in a 2x4 grid, each wave = 64x32
// (4x2 WMMA tiles -> 8 WMMAs per k-step). Double-buffered LDS fed by
// async global->LDS DMA; s_wait_asynccnt + barrier is the handoff.
// ---------------------------------------------------------------------------
template <bool OUT_BF16>
__global__ __launch_bounds__(256)
void gemm_bf16_kernel(const ushort_t* __restrict__ A,
                      const ushort_t* __restrict__ W,
                      const float* __restrict__ resid,
                      void* __restrict__ Cout,
                      int M, int N, int K) {
  __shared__ ushort_t As[2][128][40];   // +8 pad: 80B row stride, 16B aligned
  __shared__ ushort_t Bs[2][128][40];
  const int tid  = threadIdx.x;
  const int lane = tid & 31, wv = tid >> 5;
  const int wr = wv & 1, wc = wv >> 1;          // 2 row x 4 col wave grid
  const int mrow = lane & 15, half = lane >> 4;
  const int m0 = blockIdx.y * 128;
  const int n0 = blockIdx.x * 128;
  const int nk = K >> 5;

  const int ldr = tid >> 2, ldq = tid & 3;      // cooperative loader mapping

  v8f acc[4][2];
#pragma unroll
  for (int rt = 0; rt < 4; ++rt)
#pragma unroll
    for (int ct = 0; ct < 2; ++ct) acc[rt][ct] = (v8f){};

  auto issue_tile = [&](int kb, int buf) {
    const ushort_t* Ag = A + (size_t)(m0 + ldr) * K + kb * 32 + ldq * 8;
    async_load_b128(Ag,                   &As[buf][ldr][ldq * 8]);
    async_load_b128(Ag + (size_t)64 * K,  &As[buf][ldr + 64][ldq * 8]);
    const ushort_t* Wg = W + (size_t)(n0 + ldr) * K + kb * 32 + ldq * 8;
    async_load_b128(Wg,                   &Bs[buf][ldr][ldq * 8]);
    async_load_b128(Wg + (size_t)64 * K,  &Bs[buf][ldr + 64][ldq * 8]);
  };

  issue_tile(0, 0);
  wait_async0();
  __syncthreads();

  for (int kb = 0; kb < nk; ++kb) {
    const int buf = kb & 1;
    if (kb + 1 < nk) issue_tile(kb + 1, buf ^ 1);

    ushort_t (*Ac)[40] = As[buf];
    ushort_t (*Bc)[40] = Bs[buf];

    Frag16 af[4], bfr[2];
#pragma unroll
    for (int t = 0; t < 4; ++t)
#pragma unroll
      for (int j = 0; j < 8; ++j)
        af[t].u[j] = *(const unsigned*)&Ac[wr * 64 + t * 16 + mrow][kmapA(j, half)];
#pragma unroll
    for (int t = 0; t < 2; ++t)
#pragma unroll
      for (int j = 0; j < 8; ++j)
        bfr[t].u[j] = *(const unsigned*)&Bc[wc * 32 + t * 16 + mrow][16 * half + 2 * j];

#pragma unroll
    for (int rt = 0; rt < 4; ++rt)
#pragma unroll
      for (int ct = 0; ct < 2; ++ct)
        acc[rt][ct] = __builtin_amdgcn_wmma_f32_16x16x32_bf16(
            false, af[rt].v, false, bfr[ct].v, (short)0, acc[rt][ct], false, false);

    if (kb + 1 < nk) {
      wait_async0();
      __syncthreads();
    }
  }

#pragma unroll
  for (int rt = 0; rt < 4; ++rt) {
#pragma unroll
    for (int ct = 0; ct < 2; ++ct) {
      const int gcol = n0 + wc * 32 + ct * 16 + mrow;
#pragma unroll
      for (int r = 0; r < 8; ++r) {
        const int grow = m0 + wr * 64 + rt * 16 + r + 8 * half;
        float v = acc[rt][ct][r];
        if (resid) v += resid[(size_t)grow * N + gcol];
        if (OUT_BF16)
          ((ushort_t*)Cout)[(size_t)grow * N + gcol] = f2bf(v);
        else
          ((float*)Cout)[(size_t)grow * N + gcol] = v;
      }
    }
  }
}

// ---------------------------------------------------------------------------
// RoPE (+head reshape): [B,S,1024] bf16 -> [B*H, S, 64] bf16
// ---------------------------------------------------------------------------
__global__ void rope_kernel(const ushort_t* __restrict__ src,
                            ushort_t* __restrict__ dst, int rotate) {
  const int idx = blockIdx.x * blockDim.x + threadIdx.x;  // B*S*H*32 threads
  const int p = idx & 31;
  const int h = (idx >> 5) & 15;
  const int s = (idx >> 9) & 2047;
  const int b = idx >> 20;
  const size_t srow = ((size_t)(b * SEQ + s)) * D_MODEL + h * DKH + 2 * p;
  const float x0 = bf2f(src[srow]);
  const float x1 = bf2f(src[srow + 1]);
  float r0, r1;
  if (rotate) {
    const float inv = __powf(10000.0f, -(float)(2 * p) * (1.0f / DKH));
    const float ang = (float)s * inv;
    float sn, cs;
    __sincosf(ang, &sn, &cs);
    r0 = cs * x0 - sn * x1;
    r1 = sn * x0 + cs * x1;
  } else {
    r0 = x0; r1 = x1;
  }
  const size_t drow = ((size_t)(b * N_HEADS + h) * SEQ + s) * DKH + 2 * p;
  dst[drow]     = f2bf(r0);
  dst[drow + 1] = f2bf(r1);
}

// ---------------------------------------------------------------------------
// Flash-style causal attention, bf16 WMMA for Q*K^T and P*V.
// grid = (S/128, B*H), 8 waves; each wave owns 16 query rows.
// K tile staged via async DMA; V tile transposed through VALU.
// ---------------------------------------------------------------------------
__global__ __launch_bounds__(256)
void attn_kernel(const ushort_t* __restrict__ Q,
                 const ushort_t* __restrict__ Kg,
                 const ushort_t* __restrict__ Vg,
                 ushort_t* __restrict__ AO) {
  __shared__ ushort_t Kt[64][72];        // [key][d]   row-major
  __shared__ ushort_t VtT[64][72];       // [d][key]   transposed
  __shared__ ushort_t Pst[8][16][72];    // per-wave P staging [m][key]
  const int tid = threadIdx.x, lane = tid & 31, wv = tid >> 5;
  const int mrow = lane & 15, half = lane >> 4;
  const int qb = blockIdx.x, bh = blockIdx.y;
  const int b = bh >> 4, h = bh & 15;
  const size_t base = (size_t)bh * SEQ * DKH;

  // Q A-fragments (D_K=64 -> two 16x32 fragments), straight from global.
  Frag16 qf[2];
  {
    const int qrow = qb * 128 + wv * 16 + mrow;
    const ushort_t* Qrow = Q + base + (size_t)qrow * DKH;
#pragma unroll
    for (int kf = 0; kf < 2; ++kf)
#pragma unroll
      for (int j = 0; j < 8; ++j)
        qf[kf].u[j] = *(const unsigned*)(Qrow + kf * 32 + kmapA(j, half));
  }

  v8f o0 = {}, o1 = {}, o2 = {}, o3 = {};
  float mSt[8], lSt[8];
#pragma unroll
  for (int r = 0; r < 8; ++r) { mSt[r] = -3.0e38f; lSt[r] = 0.0f; }

  const int nkb = 2 * qb + 2;   // causal: keys up to end of this query block
  for (int kb = 0; kb < nkb; ++kb) {
    // K tile: async DMA straight into LDS. V tile: VALU transpose.
#pragma unroll
    for (int i = 0; i < 2; ++i) {
      const int idx = tid + i * 256;
      const int row = idx >> 3, q = idx & 7;
      async_load_b128(Kg + base + (size_t)(kb * 64 + row) * DKH + q * 8,
                      &Kt[row][q * 8]);
      const uint4 vv = *(const uint4*)(Vg + base + (size_t)(kb * 64 + row) * DKH + q * 8);
      const unsigned vw[4] = {vv.x, vv.y, vv.z, vv.w};
#pragma unroll
      for (int e = 0; e < 4; ++e) {
        VtT[q * 8 + 2 * e][row]     = (ushort_t)(vw[e] & 0xFFFFu);
        VtT[q * 8 + 2 * e + 1][row] = (ushort_t)(vw[e] >> 16);
      }
    }
    wait_async0();
    __syncthreads();

    // ---- scores: S = Q * K^T  (4 tiles of 16x16, each 2 WMMAs over d) ----
    v8f st[4];
#pragma unroll
    for (int nt = 0; nt < 4; ++nt) {
      Frag16 bk0, bk1;
#pragma unroll
      for (int j = 0; j < 8; ++j) {
        const int k0 = 16 * half + 2 * j;
        bk0.u[j] = *(const unsigned*)&Kt[nt * 16 + mrow][k0];
        bk1.u[j] = *(const unsigned*)&Kt[nt * 16 + mrow][32 + k0];
      }
      v8f z = {};
      z = __builtin_amdgcn_wmma_f32_16x16x32_bf16(false, qf[0].v, false, bk0.v, (short)0, z, false, false);
      z = __builtin_amdgcn_wmma_f32_16x16x32_bf16(false, qf[1].v, false, bk1.v, (short)0, z, false, false);
      st[nt] = z;
    }

    // ---- scale + causal mask + online softmax (16-lane shfl reductions) ----
#pragma unroll
    for (int r = 0; r < 8; ++r) {
      const int qrow = qb * 128 + wv * 16 + r + 8 * half;
      float rowm = -3.0e38f;
#pragma unroll
      for (int nt = 0; nt < 4; ++nt) {
        const int key = kb * 64 + nt * 16 + mrow;
        float v = st[nt][r] * 0.125f;            // 1/sqrt(64)
        v = (key <= qrow) ? v : -3.0e38f;
        st[nt][r] = v;
        rowm = fmaxf(rowm, v);
      }
#pragma unroll
      for (int off = 1; off < 16; off <<= 1)
        rowm = fmaxf(rowm, __shfl_xor(rowm, off, 32));
      const float mNew = fmaxf(mSt[r], rowm);
      const float corr = __expf(mSt[r] - mNew);
      float psum = 0.0f;
#pragma unroll
      for (int nt = 0; nt < 4; ++nt) {
        const float p = __expf(st[nt][r] - mNew);
        st[nt][r] = p;
        psum += p;
      }
#pragma unroll
      for (int off = 1; off < 16; off <<= 1)
        psum += __shfl_xor(psum, off, 32);
      lSt[r] = lSt[r] * corr + psum;
      mSt[r] = mNew;
      o0[r] = o0[r] * corr; o1[r] = o1[r] * corr;
      o2[r] = o2[r] * corr; o3[r] = o3[r] * corr;
    }

    // ---- stage P (D-layout -> A-layout round trip; same-wave LDS is in-order)
#pragma unroll
    for (int nt = 0; nt < 4; ++nt)
#pragma unroll
      for (int r = 0; r < 8; ++r)
        Pst[wv][r + 8 * half][nt * 16 + mrow] = f2bf(st[nt][r]);

    Frag16 pf0, pf1;
#pragma unroll
    for (int j = 0; j < 8; ++j) {
      const int ka = kmapA(j, half);
      pf0.u[j] = *(const unsigned*)&Pst[wv][mrow][ka];
      pf1.u[j] = *(const unsigned*)&Pst[wv][mrow][32 + ka];
    }

    // ---- O += P * V ----
#pragma unroll
    for (int nt = 0; nt < 4; ++nt) {
      Frag16 bv0, bv1;
#pragma unroll
      for (int j = 0; j < 8; ++j) {
        const int k0 = 16 * half + 2 * j;
        bv0.u[j] = *(const unsigned*)&VtT[nt * 16 + mrow][k0];
        bv1.u[j] = *(const unsigned*)&VtT[nt * 16 + mrow][32 + k0];
      }
      v8f o = (nt == 0) ? o0 : ((nt == 1) ? o1 : ((nt == 2) ? o2 : o3));
      o = __builtin_amdgcn_wmma_f32_16x16x32_bf16(false, pf0.v, false, bv0.v, (short)0, o, false, false);
      o = __builtin_amdgcn_wmma_f32_16x16x32_bf16(false, pf1.v, false, bv1.v, (short)0, o, false, false);
      if (nt == 0) o0 = o; else if (nt == 1) o1 = o; else if (nt == 2) o2 = o; else o3 = o;
    }
    __syncthreads();
  }

  // ---- epilogue: O /= l, scatter back to [B,S,1024] head-interleaved ----
#pragma unroll
  for (int r = 0; r < 8; ++r) {
    const int qrow = qb * 128 + wv * 16 + r + 8 * half;
    const float inv = 1.0f / lSt[r];
    const size_t orow = ((size_t)(b * SEQ) + qrow) * D_MODEL + h * DKH;
    AO[orow +  0 + mrow] = f2bf(o0[r] * inv);
    AO[orow + 16 + mrow] = f2bf(o1[r] * inv);
    AO[orow + 32 + mrow] = f2bf(o2[r] * inv);
    AO[orow + 48 + mrow] = f2bf(o3[r] * inv);
  }
}

// ---------------------------------------------------------------------------
// SwiGLU gate: g = silu(a) * u  (bf16 in/out)
// ---------------------------------------------------------------------------
__global__ void silu_mul_kernel(const ushort_t* __restrict__ a,
                                const ushort_t* __restrict__ u,
                                ushort_t* __restrict__ g, int n) {
  const int i = blockIdx.x * blockDim.x + threadIdx.x;
  if (i < n) {
    const float af = bf2f(a[i]);
    const float uf = bf2f(u[i]);
    const float s = af / (1.0f + __expf(-af));
    g[i] = f2bf(s * uf);
  }
}

// ---------------------------------------------------------------------------
// Host orchestration
// ---------------------------------------------------------------------------
extern "C" void kernel_launch(void* const* d_in, const int* in_sizes, int n_in,
                              void* d_out, int out_size, void* d_ws, size_t ws_size,
                              hipStream_t stream) {
  (void)in_sizes; (void)n_in; (void)out_size; (void)ws_size;
  const float* x  = (const float*)d_in[0];
  const float* wq = (const float*)d_in[1];
  const float* wk = (const float*)d_in[2];
  const float* wv = (const float*)d_in[3];
  const float* wo = (const float*)d_in[4];
  const float* w1 = (const float*)d_in[5];
  const float* w2 = (const float*)d_in[6];
  const float* w3 = (const float*)d_in[7];
  const float* g1 = (const float*)d_in[8];
  const float* g2 = (const float*)d_in[9];

  char* ws = (char*)d_ws;
  size_t off = 0;
  auto alloc = [&](size_t bytes) -> void* {
    void* p = ws + off;
    off += (bytes + 255) & ~(size_t)255;
    return p;
  };

  const size_t MM = (size_t)D_MODEL * D_MODEL;        // 1M
  const size_t MF = (size_t)D_FF * D_MODEL;           // 4M
  ushort_t* wqb = (ushort_t*)alloc(MM * 2);
  ushort_t* wkb = (ushort_t*)alloc(MM * 2);
  ushort_t* wvb = (ushort_t*)alloc(MM * 2);
  ushort_t* wob = (ushort_t*)alloc(MM * 2);
  ushort_t* w1b = (ushort_t*)alloc(MF * 2);
  ushort_t* w3b = (ushort_t*)alloc(MF * 2);
  ushort_t* w2b = (ushort_t*)alloc(MF * 2);
  ushort_t* hb  = (ushort_t*)alloc((size_t)NTOK * D_MODEL * 2);
  float*    x2f = (float*)   alloc((size_t)NTOK * D_MODEL * 4);
  ushort_t* qt  = (ushort_t*)alloc((size_t)NTOK * D_MODEL * 2);
  ushort_t* kt  = (ushort_t*)alloc((size_t)NTOK * D_MODEL * 2);
  ushort_t* vt  = (ushort_t*)alloc((size_t)NTOK * D_MODEL * 2);
  ushort_t* qh  = (ushort_t*)alloc((size_t)NTOK * D_MODEL * 2);
  ushort_t* kh  = (ushort_t*)alloc((size_t)NTOK * D_MODEL * 2);
  ushort_t* vh  = (ushort_t*)alloc((size_t)NTOK * D_MODEL * 2);
  ushort_t* ao  = (ushort_t*)alloc((size_t)NTOK * D_MODEL * 2);
  (void)alloc((size_t)NTOK * D_MODEL * 2);            // pad for ub alias
  // FFN activations alias dead attention buffers (qt..qh = 32MB, kh..pad = 32MB)
  ushort_t* ab = qt;
  ushort_t* ub = kh;

  // 1) weights -> bf16
  cvt_bf16_kernel<<<(int)(MM / 256), 256, 0, stream>>>(wq, wqb, (int)MM);
  cvt_bf16_kernel<<<(int)(MM / 256), 256, 0, stream>>>(wk, wkb, (int)MM);
  cvt_bf16_kernel<<<(int)(MM / 256), 256, 0, stream>>>(wv, wvb, (int)MM);
  cvt_bf16_kernel<<<(int)(MM / 256), 256, 0, stream>>>(wo, wob, (int)MM);
  cvt_bf16_kernel<<<(int)(MF / 256), 256, 0, stream>>>(w1, w1b, (int)MF);
  cvt_bf16_kernel<<<(int)(MF / 256), 256, 0, stream>>>(w3, w3b, (int)MF);
  cvt_bf16_kernel<<<(int)(MF / 256), 256, 0, stream>>>(w2, w2b, (int)MF);

  // 2) rmsnorm(x, g1)
  rmsnorm_kernel<<<NTOK, 256, 0, stream>>>(x, g1, hb);

  // 3) QKV projections (block tile 128x128)
  const dim3 gProj(D_MODEL / 128, NTOK / 128);        // (8, 32)
  gemm_bf16_kernel<true><<<gProj, 256, 0, stream>>>(hb, wqb, nullptr, qt, NTOK, D_MODEL, D_MODEL);
  gemm_bf16_kernel<true><<<gProj, 256, 0, stream>>>(hb, wkb, nullptr, kt, NTOK, D_MODEL, D_MODEL);
  gemm_bf16_kernel<true><<<gProj, 256, 0, stream>>>(hb, wvb, nullptr, vt, NTOK, D_MODEL, D_MODEL);

  // 4) RoPE + head reshape (V is reshape-only)
  const int ropeN = BSZ * SEQ * N_HEADS * 32;         // 2,097,152 threads
  rope_kernel<<<ropeN / 256, 256, 0, stream>>>(qt, qh, 1);
  rope_kernel<<<ropeN / 256, 256, 0, stream>>>(kt, kh, 1);
  rope_kernel<<<ropeN / 256, 256, 0, stream>>>(vt, vh, 0);

  // 5) attention
  attn_kernel<<<dim3(SEQ / 128, BSZ * N_HEADS), 256, 0, stream>>>(qh, kh, vh, ao);

  // 6) output projection + residual -> x2
  gemm_bf16_kernel<false><<<gProj, 256, 0, stream>>>(ao, wob, x, x2f, NTOK, D_MODEL, D_MODEL);

  // 7) rmsnorm(x2, g2)
  rmsnorm_kernel<<<NTOK, 256, 0, stream>>>(x2f, g2, hb);

  // 8) FFN up projections
  const dim3 gFF(D_FF / 128, NTOK / 128);             // (32, 32)
  gemm_bf16_kernel<true><<<gFF, 256, 0, stream>>>(hb, w1b, nullptr, ab, NTOK, D_FF, D_MODEL);
  gemm_bf16_kernel<true><<<gFF, 256, 0, stream>>>(hb, w3b, nullptr, ub, NTOK, D_FF, D_MODEL);

  // 9) SwiGLU gate
  const int ffn = NTOK * D_FF;                        // 16,777,216
  silu_mul_kernel<<<ffn / 256, 256, 0, stream>>>(ab, ub, ub, ffn);

  // 10) down projection + residual -> out (f32)
  gemm_bf16_kernel<false><<<gProj, 256, 0, stream>>>(ub, w2b, x2f, d_out, NTOK, D_MODEL, D_FF);
}
